// GraphConv_79250736545937
// MI455X (gfx1250) — compile-verified, matrix operand order checked by reading, not run on hardware
//
#include <hip/hip_runtime.h>

#define N_NODES 50000
#define N_EDGES 800000
#define FEATS   128

typedef __attribute__((ext_vector_type(16))) __bf16 v16bf;
typedef __attribute__((ext_vector_type(8)))  float  v8f;

// Workspace layout (bytes)
#define DEG_OUT_OFF 0
#define DEG_IN_OFF  (256 * 1024)
#define AGGB_OFF    (512 * 1024)                       // bf16 agg: 50000*128*2 = 12.8 MB
#define WT_OFF      (AGGB_OFF + N_NODES * FEATS * 2)   // bf16 W^T: 32 KB
// total ~13.4 MB of d_ws used

__device__ __forceinline__ unsigned short f2bf(float f) {
    unsigned u = __float_as_uint(f);
    unsigned r = u + 0x7fffu + ((u >> 16) & 1u);   // round-to-nearest-even
    return (unsigned short)(r >> 16);
}

// ---- 1) zero agg (lives in d_out) and both degree arrays ----------------
__global__ void k_init(float4* __restrict__ out4,
                       float* __restrict__ deg_out,
                       float* __restrict__ deg_in) {
    int i = blockIdx.x * blockDim.x + threadIdx.x;
    int stride = gridDim.x * blockDim.x;
    const float4 z = make_float4(0.f, 0.f, 0.f, 0.f);
    for (int j = i; j < (N_NODES * FEATS) / 4; j += stride) out4[j] = z;
    for (int j = i; j < N_NODES; j += stride) { deg_out[j] = 0.f; deg_in[j] = 0.f; }
}

// ---- 2) degree histograms ----------------------------------------------
__global__ void k_degree(const int* __restrict__ src, const int* __restrict__ dst,
                         float* __restrict__ deg_out, float* __restrict__ deg_in) {
    int e = blockIdx.x * blockDim.x + threadIdx.x;
    if (e >= N_EDGES) return;
    unsafeAtomicAdd(&deg_out[src[e]], 1.0f);
    unsafeAtomicAdd(&deg_in[dst[e]], 1.0f);
}

// ---- 3) SpMM: agg[dst] += feat[src] * deg_out[src]^-0.5 -----------------
// One wave32 per edge; lane handles 4 consecutive features (float4 gather).
__global__ void __launch_bounds__(256) k_spmm(const float* __restrict__ feat,
                                              const int* __restrict__ src,
                                              const int* __restrict__ dst,
                                              const float* __restrict__ deg_out,
                                              float* __restrict__ agg) {
    unsigned t = blockIdx.x * blockDim.x + threadIdx.x;
    unsigned e = t >> 5;
    if (e >= N_EDGES) return;
    int lane = t & 31;
    int s = src[e];
    int d = dst[e];
    float sc = rsqrtf(fmaxf(deg_out[s], 1.0f));
    const float4 v = *(const float4*)(feat + (size_t)s * FEATS + lane * 4);
    float* p = agg + (size_t)d * FEATS + lane * 4;
    unsafeAtomicAdd(p + 0, v.x * sc);
    unsafeAtomicAdd(p + 1, v.y * sc);
    unsafeAtomicAdd(p + 2, v.z * sc);
    unsafeAtomicAdd(p + 3, v.w * sc);
}

// ---- 4) convert agg fp32 -> bf16 (packed pairs) -------------------------
__global__ void k_cvt_agg(const float* __restrict__ agg, unsigned* __restrict__ aggb) {
    int i = blockIdx.x * blockDim.x + threadIdx.x;
    if (i >= (N_NODES * FEATS) / 2) return;
    float a0 = agg[2 * i];
    float a1 = agg[2 * i + 1];
    aggb[i] = (unsigned)f2bf(a0) | ((unsigned)f2bf(a1) << 16);
}

// ---- 5) convert weight fp32 -> bf16, transposed (wt[n][k] = w[k][n]) ----
__global__ void k_cvt_w(const float* __restrict__ w, unsigned short* __restrict__ wt) {
    int i = blockIdx.x * blockDim.x + threadIdx.x;
    if (i >= FEATS * FEATS) return;
    int k = i >> 7;
    int n = i & 127;
    wt[n * FEATS + k] = f2bf(w[i]);
}

// ---- 6) WMMA GEMM: out = (agg @ W) * deg_in^-0.5 + bias -----------------
// Block = 256 threads = 8 waves; block b handles rows [16b,16b+16), wave w
// handles columns [16w, 16w+16). K=128 -> 4 x v_wmma_f32_16x16x32_bf16.
__global__ void __launch_bounds__(256) k_gemm(const unsigned short* __restrict__ aggb,
                                              const unsigned short* __restrict__ wt,
                                              const float* __restrict__ deg_in,
                                              const float* __restrict__ bias,
                                              float* __restrict__ out) {
    int lane  = threadIdx.x & 31;
    int tileN = threadIdx.x >> 5;   // 0..7
    int tileM = blockIdx.x;         // 0..3124
    int hi = lane >> 4;             // lane group (K offset select)
    int ml = lane & 15;             // A-row / B-col / C-col within tile

    int arow = tileM * 16 + ml;     // A fragment row
    int bcol = tileN * 16 + ml;     // B fragment column (wt is [n][k])

    v8f acc = {0.f, 0.f, 0.f, 0.f, 0.f, 0.f, 0.f, 0.f};

    #pragma unroll
    for (int s = 0; s < 4; ++s) {
        int kb = s * 32 + hi * 8;   // ISA A/B 16-bit layout: VGPR0-3 = K kb..kb+7,
                                    //                        VGPR4-7 = K kb+16..kb+23
        union { uint4 q[2]; v16bf v; } A, B;
        A.q[0] = *(const uint4*)(aggb + (size_t)arow * FEATS + kb);
        A.q[1] = *(const uint4*)(aggb + (size_t)arow * FEATS + kb + 16);
        B.q[0] = *(const uint4*)(wt   + (size_t)bcol * FEATS + kb);
        B.q[1] = *(const uint4*)(wt   + (size_t)bcol * FEATS + kb + 16);
        acc = __builtin_amdgcn_wmma_f32_16x16x32_bf16(
                  false, A.v, false, B.v, (short)0, acc, false, false);
    }

    int n = tileN * 16 + ml;        // C layout: lanes 0-15 N=lane, lanes 16-31 N=lane-16
    float b = bias[n];
    #pragma unroll
    for (int r = 0; r < 8; ++r) {
        int m = tileM * 16 + hi * 8 + r;   // C layout: VGPR r -> M = r (+8 for hi lanes)
        float sc = rsqrtf(fmaxf(deg_in[m], 1.0f));
        out[(size_t)m * FEATS + n] = acc[r] * sc + b;
    }
}

extern "C" void kernel_launch(void* const* d_in, const int* in_sizes, int n_in,
                              void* d_out, int out_size, void* d_ws, size_t ws_size,
                              hipStream_t stream) {
    const float* feat = (const float*)d_in[0];
    const int*   src  = (const int*)d_in[1];
    const int*   dst  = (const int*)d_in[2];
    const float* w    = (const float*)d_in[3];
    const float* bias = (const float*)d_in[4];
    float* out = (float*)d_out;

    char* ws = (char*)d_ws;
    float*          deg_out = (float*)(ws + DEG_OUT_OFF);
    float*          deg_in  = (float*)(ws + DEG_IN_OFF);
    unsigned short* aggb    = (unsigned short*)(ws + AGGB_OFF);
    unsigned short* wt      = (unsigned short*)(ws + WT_OFF);

    // 1) zero accumulators (agg scratch lives in d_out)
    k_init<<<2048, 256, 0, stream>>>((float4*)out, deg_out, deg_in);
    // 2) degrees
    k_degree<<<(N_EDGES + 255) / 256, 256, 0, stream>>>(src, dst, deg_out, deg_in);
    // 3) SpMM scatter-add (one wave per edge)
    k_spmm<<<(N_EDGES * 32) / 256, 256, 0, stream>>>(feat, src, dst, deg_out, out);
    // 4) agg -> bf16
    k_cvt_agg<<<((N_NODES * FEATS / 2) + 255) / 256, 256, 0, stream>>>(out, (unsigned*)aggb);
    // 5) W -> bf16 transposed
    k_cvt_w<<<(FEATS * FEATS + 255) / 256, 256, 0, stream>>>(w, wt);
    // 6) WMMA GEMM + fused epilogue (overwrites d_out with final result)
    k_gemm<<<N_NODES / 16, 256, 0, stream>>>(aggb, wt, deg_in, bias, out);
}
